// MPMA_24309514895627
// MI455X (gfx1250) — matrix operand order repaired and checked by reading, not compiled, not run on hardware
//
#include <hip/hip_runtime.h>
#include <hip/hip_bf16.h>

typedef __attribute__((ext_vector_type(16))) _Float16 v16h;
typedef __attribute__((ext_vector_type(8)))  float    v8f;

#define LN_EPS 1e-5f

static __device__ __forceinline__ float siluf(float x) {
  return x * (1.0f / (1.0f + __expf(-x)));
}
static __device__ __forceinline__ float softplusf(float x) {
  return (x > 20.0f) ? x : __logf(1.0f + __expf(x));
}

// ==================== async global -> LDS weight staging =====================
// Each enabled lane issues a 16-byte async copy (GLOBAL_LOAD_ASYNC_TO_LDS_B128,
// tracked by ASYNCcnt). lds_off is an addrspace(3) byte offset, gsrc must be
// 16B aligned, nfloats a multiple of 4. Followed by s_wait_asynccnt 0; caller
// must __syncthreads() afterwards.
static __device__ __forceinline__ void stage_async(uint32_t lds_off,
                                                   const float* __restrict__ gsrc,
                                                   int nfloats) {
  for (int i = threadIdx.x * 4; i < nfloats; i += blockDim.x * 4) {
    uint32_t l = lds_off + (uint32_t)i * 4u;
    uint64_t g = (uint64_t)(uintptr_t)(gsrc + i);
    asm volatile("global_load_async_to_lds_b128 %0, %1, off"
                 :: "v"(l), "v"(g) : "memory");
  }
  asm volatile("s_wait_asynccnt 0" ::: "memory");
}

// ======================= WMMA fragment helpers (wave32) =======================
// A fragment: 16x32 f16. Lane L<16 -> M=L, K in {0..7, 16..23};
// lane L>=16 -> M=L-16, K in {8..15, 24..31}.  Element i of the v16h maps to
// k = kbase + i (i<8) and k = 16 + kbase + (i-8) (i>=8), i.e. two contiguous
// 8-float runs of the source row -> four float4 vector loads.
template <int LD, int KVALID>
static __device__ __forceinline__ v16h load_a_frag(const float* __restrict__ src) {
  const int lane  = threadIdx.x & 31;
  const int m     = lane & 15;
  const int kbase = (lane < 16) ? 0 : 8;
  const float* row = src + m * LD + kbase;
  const float4 q0 = *reinterpret_cast<const float4*>(row);
  const float4 q1 = *reinterpret_cast<const float4*>(row + 4);
  float4 q2, q3;
  if constexpr (KVALID > 16) {
    q2 = *reinterpret_cast<const float4*>(row + 16);
    q3 = *reinterpret_cast<const float4*>(row + 20);
  } else {
    q2 = make_float4(0.f, 0.f, 0.f, 0.f);
    q3 = make_float4(0.f, 0.f, 0.f, 0.f);
  }
  v16h a;
  a[0]  = (_Float16)q0.x; a[1]  = (_Float16)q0.y;
  a[2]  = (_Float16)q0.z; a[3]  = (_Float16)q0.w;
  a[4]  = (_Float16)q1.x; a[5]  = (_Float16)q1.y;
  a[6]  = (_Float16)q1.z; a[7]  = (_Float16)q1.w;
  a[8]  = (_Float16)q2.x; a[9]  = (_Float16)q2.y;
  a[10] = (_Float16)q2.z; a[11] = (_Float16)q2.w;
  a[12] = (_Float16)q3.x; a[13] = (_Float16)q3.y;
  a[14] = (_Float16)q3.z; a[15] = (_Float16)q3.w;
  return a;
}

// B fragment: 32x16 f16. Lane L -> column N=L%16; K = (L<16?0:8) + i (i<8), 16 + (L<16?0:8) + (i-8) (i>=8).
// Source row-major f32 [K][N] (global or LDS).  Padding handled branch-free:
// indices clamped in-bounds, value selected to zero (v_cndmask).
template <int LD, int KVALID, int NVALID>
static __device__ __forceinline__ v16h load_b_frag(const float* w) {
  const int lane  = threadIdx.x & 31;
  const int n     = lane & 15;
  const int kbase = (lane < 16) ? 0 : 8;
  const bool nok  = (NVALID >= 16) || (n < NVALID);
  const int  nc   = nok ? n : 0;
  v16h b;
#pragma unroll
  for (int i = 0; i < 16; ++i) {
    const int  k   = kbase + i + ((i >= 8) ? 8 : 0);
    const bool kok = (KVALID >= 32) || (k < KVALID);
    const int  kc  = kok ? k : 0;
    float f = w[kc * LD + nc];
    if constexpr (KVALID < 32 || NVALID < 16) {
      f = (kok && nok) ? f : 0.0f;
    }
    b[i] = (_Float16)f;
  }
  return b;
}

// C/D fragment: 16x16 f32, lane L -> N=L%16, VGPR v -> M = v + (L<16?0:8).
template <int LD>
static __device__ __forceinline__ void store_d_frag(float* __restrict__ dst, v8f d) {
  const int lane  = threadIdx.x & 31;
  const int n     = lane & 15;
  const int mbase = (lane < 16) ? 0 : 8;
#pragma unroll
  for (int v = 0; v < 8; ++v) dst[(mbase + v) * LD + n] = d[v];
}

template <int LD>
static __device__ __forceinline__ v8f load_c_frag_scaled(const float* __restrict__ src,
                                                         float scale) {
  const int lane  = threadIdx.x & 31;
  const int n     = lane & 15;
  const int mbase = (lane < 16) ? 0 : 8;
  v8f c;
#pragma unroll
  for (int v = 0; v < 8; ++v) c[v] = scale * src[(mbase + v) * LD + n];
  return c;
}

static __device__ __forceinline__ v8f load_bias_frag(const float* __restrict__ bias) {
  const float bv = bias[threadIdx.x & 15];
  v8f c;
#pragma unroll
  for (int v = 0; v < 8; ++v) c[v] = bv;
  return c;
}

static __device__ __forceinline__ v8f wmma_f16(v16h a, v16h b, v8f c) {
  return __builtin_amdgcn_wmma_f32_16x16x32_f16(false, a, false, b, (short)0, c,
                                                false, false);
}

// ============================ pipeline kernels ================================

// (B,C,H,W) f32 -> (B, H*W, C)
__global__ void k_nchw_to_nhwc(const float* __restrict__ x, float* __restrict__ xh,
                               int B, int C, int H, int W) {
  long idx = (long)blockIdx.x * blockDim.x + threadIdx.x;
  long total = (long)B * C * H * W;
  if (idx >= total) return;
  int w = (int)(idx % W);
  int h = (int)((idx / W) % H);
  int c = (int)((idx / ((long)W * H)) % C);
  int b = (int)(idx / ((long)W * H * C));
  xh[(((long)b * H + h) * W + w) * C + c] = x[idx];
}

// 2x2 stride-2 down conv: out[b,i,j,o] = sum_{c,p,q} in[b,2i+p,2j+q,c]*dw[o,c,p,q]
__global__ void k_down(const float* __restrict__ in, const float* __restrict__ dw,
                       float* __restrict__ out, int B, int H, int W) {
  const int Ho = H / 2, Wo = W / 2;
  long idx = (long)blockIdx.x * blockDim.x + threadIdx.x;
  long total = (long)B * Ho * Wo * 64;
  if (idx >= total) return;
  int o = (int)(idx & 63);
  int j = (int)((idx >> 6) % Wo);
  int i = (int)((idx >> 6) / Wo % Ho);
  int b = (int)(idx / ((long)Ho * Wo * 64));
  float acc = 0.0f;
#pragma unroll
  for (int p = 0; p < 2; ++p) {
#pragma unroll
    for (int q = 0; q < 2; ++q) {
      const float* src = &in[(((long)b * H + 2 * i + p) * W + (2 * j + q)) * 64];
      const float* wv  = &dw[(long)o * 64 * 4 + p * 2 + q];
      for (int c = 0; c < 64; ++c) acc += src[c] * wv[c * 4];
    }
  }
  out[idx] = acc;
}

// LayerNorm over C=64 then split into 4 groups of 16 -> xc[(g*B+b)*n + t][d]
__global__ void k_ln_split(const float* __restrict__ xh, const float* __restrict__ lnw,
                           const float* __restrict__ lnb, float* __restrict__ xc,
                           int B, int n) {
  long t = (long)blockIdx.x * blockDim.x + threadIdx.x;
  if (t >= (long)B * n) return;
  int b  = (int)(t / n);
  int ts = (int)(t % n);
  const float* src = &xh[t * 64];
  float val[64];
  float m = 0.0f;
#pragma unroll
  for (int c = 0; c < 64; ++c) { val[c] = src[c]; m += val[c]; }
  m *= (1.0f / 64.0f);
  float var = 0.0f;
#pragma unroll
  for (int c = 0; c < 64; ++c) { float d = val[c] - m; var += d * d; }
  var *= (1.0f / 64.0f);
  const float inv = rsqrtf(var + LN_EPS);
#pragma unroll
  for (int g = 0; g < 4; ++g) {
#pragma unroll
    for (int d = 0; d < 16; ++d) {
      const int c = g * 16 + d;
      xc[(((long)(g * B + b) * n + ts) * 16) + d] = (val[c] - m) * inv * lnw[c] + lnb[c];
    }
  }
}

// in_proj: xc (T x 16) @ w (16 x 64) -> xm_raw (T x 32) | z (T x 32).
// Weights staged to LDS via async copy; WMMA K padded 16->32.
__global__ void k_inproj(const float* __restrict__ xc, const float* __restrict__ w,
                         float* __restrict__ xm_raw, float* __restrict__ z, int ntiles) {
  __shared__ __align__(16) float wl[16 * 64];
  stage_async((uint32_t)(uintptr_t)&wl[0], w, 16 * 64);
  __syncthreads();
  const int wave = (int)(((long)blockIdx.x * blockDim.x + threadIdx.x) >> 5);
  if (wave >= ntiles) return;
  v16h a = load_a_frag<16, 16>(xc + (long)wave * 16 * 16);
  const v8f zero = {0.f, 0.f, 0.f, 0.f, 0.f, 0.f, 0.f, 0.f};
#pragma unroll
  for (int nt = 0; nt < 4; ++nt) {
    v16h b = load_b_frag<64, 16, 16>(wl + nt * 16);
    v8f acc = wmma_f16(a, b, zero);
    float* dst = (nt < 2) ? (xm_raw + (long)wave * 16 * 32 + nt * 16)
                          : (z + (long)wave * 16 * 32 + (nt - 2) * 16);
    store_d_frag<32>(dst, acc);
  }
}

// causal depthwise conv (k=4) + bias + SiLU
__global__ void k_conv_silu(const float* __restrict__ xm_raw, const float* __restrict__ cw,
                            const float* __restrict__ cb, float* __restrict__ xm,
                            long Ttot, int n) {
  long idx = (long)blockIdx.x * blockDim.x + threadIdx.x;
  if (idx >= Ttot * 32) return;
  int d   = (int)(idx & 31);
  long tt = idx >> 5;
  int tin = (int)(tt % n);
  float acc = cb[d];
#pragma unroll
  for (int k = 0; k < 4; ++k) {
    int src = tin - 3 + k;
    if (src >= 0) acc += xm_raw[(tt - 3 + k) * 32 + d] * cw[d * 4 + k];
  }
  xm[idx] = siluf(acc);
}

// x_proj: xm (T x 32) @ w (32 x 33) -> x_dbl (T x 48, cols 33..47 zero)
__global__ void k_xproj(const float* __restrict__ xm, const float* __restrict__ w,
                        float* __restrict__ x_dbl, int ntiles) {
  __shared__ __align__(16) float wl[32 * 33];
  stage_async((uint32_t)(uintptr_t)&wl[0], w, 32 * 33);
  __syncthreads();
  const int wave = (int)(((long)blockIdx.x * blockDim.x + threadIdx.x) >> 5);
  if (wave >= ntiles) return;
  v16h a = load_a_frag<32, 32>(xm + (long)wave * 16 * 32);
  const v8f zero = {0.f, 0.f, 0.f, 0.f, 0.f, 0.f, 0.f, 0.f};
  float* dst = x_dbl + (long)wave * 16 * 48;
  v16h b0 = load_b_frag<33, 32, 16>(wl);
  store_d_frag<48>(dst, wmma_f16(a, b0, zero));
  v16h b1 = load_b_frag<33, 32, 16>(wl + 16);
  store_d_frag<48>(dst + 16, wmma_f16(a, b1, zero));
  v16h b2 = load_b_frag<33, 32, 1>(wl + 32);
  store_d_frag<48>(dst + 32, wmma_f16(a, b2, zero));
}

// dt = softplus(dt_r * dt_proj_w + dt_proj_b); dt_rank == 1
__global__ void k_dt(const float* __restrict__ x_dbl, const float* __restrict__ dtw,
                     const float* __restrict__ dtb, float* __restrict__ dt, long Ttot) {
  long idx = (long)blockIdx.x * blockDim.x + threadIdx.x;
  if (idx >= Ttot * 32) return;
  int d   = (int)(idx & 31);
  long tt = idx >> 5;
  dt[idx] = softplusf(x_dbl[tt * 48] * dtw[d] + dtb[d]);
}

// selective scan: block = 1 wave; owns 2 channels x 16 states; y reduced with
// shfl_xor within each 16-lane state group. grid = (32 seqs, 16 channel-pairs)
__global__ void k_scan(const float* __restrict__ xm, const float* __restrict__ dt,
                       const float* __restrict__ x_dbl, const float* __restrict__ A_log,
                       float* __restrict__ ys, int n) {
  const int seq  = blockIdx.x;
  const int lane = threadIdx.x & 31;
  const int d    = blockIdx.y * 2 + (lane >> 4);
  const int s    = lane & 15;
  const float A  = -__expf(A_log[d * 16 + s]);
  float h = 0.0f;
  const long base = (long)seq * n;
  for (int t = 0; t < n; ++t) {
    const long tt = base + t;
    if (t + 8 < n) __builtin_prefetch(&x_dbl[(tt + 8) * 48], 0, 1);
    const float dv = dt[tt * 32 + d];
    const float xv = xm[tt * 32 + d];
    const float bv = x_dbl[tt * 48 + 1 + s];
    const float cv = x_dbl[tt * 48 + 17 + s];
    h = __expf(dv * A) * h + (dv * xv) * bv;
    float part = h * cv;
    part += __shfl_xor(part, 1, 32);
    part += __shfl_xor(part, 2, 32);
    part += __shfl_xor(part, 4, 32);
    part += __shfl_xor(part, 8, 32);
    if (s == 0) ys[tt * 32 + d] = part;
  }
}

// y = (ys + xm * D) * silu(z), in place on ys
__global__ void k_ycomb(float* __restrict__ ys, const float* __restrict__ xm,
                        const float* __restrict__ z, const float* __restrict__ Dv,
                        long Ttot) {
  long idx = (long)blockIdx.x * blockDim.x + threadIdx.x;
  if (idx >= Ttot * 32) return;
  int d = (int)(idx & 31);
  ys[idx] = (ys[idx] + xm[idx] * Dv[d]) * siluf(z[idx]);
}

// out_proj: y (T x 32) @ w (32 x 16) + skip_scale * xc -> ym (T x 16)
__global__ void k_outproj(const float* __restrict__ y, const float* __restrict__ w,
                          const float* __restrict__ xc, const float* __restrict__ skip,
                          float* __restrict__ ym, int ntiles) {
  __shared__ __align__(16) float wl[32 * 16];
  stage_async((uint32_t)(uintptr_t)&wl[0], w, 32 * 16);
  __syncthreads();
  const int wave = (int)(((long)blockIdx.x * blockDim.x + threadIdx.x) >> 5);
  if (wave >= ntiles) return;
  const float sk = skip[0];
  v16h a = load_a_frag<32, 32>(y + (long)wave * 16 * 32);
  v16h b = load_b_frag<16, 32, 16>(wl);
  v8f acc = load_c_frag_scaled<16>(xc + (long)wave * 16 * 16, sk);
  acc = wmma_f16(a, b, acc);
  store_d_frag<16>(ym + (long)wave * 16 * 16, acc);
}

// merge 4 groups back to C=64, LayerNorm -> ymn (B*n x 64)
__global__ void k_merge_ln(const float* __restrict__ ym, const float* __restrict__ lnw,
                           const float* __restrict__ lnb, float* __restrict__ ymn,
                           int B, int n) {
  long t = (long)blockIdx.x * blockDim.x + threadIdx.x;
  if (t >= (long)B * n) return;
  int b  = (int)(t / n);
  int ts = (int)(t % n);
  float val[64];
#pragma unroll
  for (int g = 0; g < 4; ++g)
#pragma unroll
    for (int d = 0; d < 16; ++d)
      val[g * 16 + d] = ym[(((long)(g * B + b) * n + ts) * 16) + d];
  float m = 0.0f;
#pragma unroll
  for (int c = 0; c < 64; ++c) m += val[c];
  m *= (1.0f / 64.0f);
  float var = 0.0f;
#pragma unroll
  for (int c = 0; c < 64; ++c) { float dd = val[c] - m; var += dd * dd; }
  var *= (1.0f / 64.0f);
  const float inv = rsqrtf(var + LN_EPS);
#pragma unroll
  for (int c = 0; c < 64; ++c)
    ymn[t * 64 + c] = (val[c] - m) * inv * lnw[c] + lnb[c];
}

// proj: ymn (BT x 64) @ w (64 x 64) + bias -> out (BT x 64). Two K=32 WMMA steps.
// 16 KB weight matrix staged into LDS once per block via async copies.
__global__ void k_proj(const float* __restrict__ ymn, const float* __restrict__ w,
                       const float* __restrict__ bias, float* __restrict__ out,
                       int ntiles) {
  __shared__ __align__(16) float wl[64 * 64];
  stage_async((uint32_t)(uintptr_t)&wl[0], w, 64 * 64);
  __syncthreads();
  const int wave = (int)(((long)blockIdx.x * blockDim.x + threadIdx.x) >> 5);
  if (wave >= ntiles) return;
  v16h a0 = load_a_frag<64, 32>(ymn + (long)wave * 16 * 64);
  v16h a1 = load_a_frag<64, 32>(ymn + (long)wave * 16 * 64 + 32);
#pragma unroll
  for (int nt = 0; nt < 4; ++nt) {
    v16h b0 = load_b_frag<64, 32, 16>(wl + nt * 16);
    v16h b1 = load_b_frag<64, 32, 16>(wl + 32 * 64 + nt * 16);
    v8f acc = load_bias_frag(bias + nt * 16);
    acc = wmma_f16(a0, b0, acc);
    acc = wmma_f16(a1, b1, acc);
    store_d_frag<64>(out + (long)wave * 16 * 64 + nt * 16, acc);
  }
}

// resn[b,2i+p,2j+q,o] = outl[...] + alpha * sum_c res[b,i,j,c] * uw[c,o,p,q]
__global__ void k_upadd(const float* __restrict__ res, const float* __restrict__ uw,
                        const float* __restrict__ outl, float alpha,
                        float* __restrict__ resn, int B, int Hs, int Ws) {
  long idx = (long)blockIdx.x * blockDim.x + threadIdx.x;
  long total = (long)B * Hs * Ws * 64;
  if (idx >= total) return;
  int o = (int)(idx & 63);
  int j = (int)((idx >> 6) % Ws);
  int i = (int)((idx >> 6) / Ws % Hs);
  int b = (int)(idx / ((long)Hs * Ws * 64));
  const float* rp = &res[(((long)b * Hs + i) * Ws + j) * 64];
  float s00 = 0.f, s01 = 0.f, s10 = 0.f, s11 = 0.f;
  for (int c = 0; c < 64; ++c) {
    const float rv = rp[c];
    const float* wv = &uw[((long)c * 64 + o) * 4];
    s00 += rv * wv[0];
    s01 += rv * wv[1];
    s10 += rv * wv[2];
    s11 += rv * wv[3];
  }
  const int Wo = 2 * Ws;
  long base = (((long)b * (2 * Hs) + 2 * i) * Wo + 2 * j) * 64 + o;
  resn[base]                      = outl[base] + alpha * s00;
  resn[base + 64]                 = outl[base + 64] + alpha * s01;
  resn[base + (long)Wo * 64]      = outl[base + (long)Wo * 64] + alpha * s10;
  resn[base + (long)Wo * 64 + 64] = outl[base + (long)Wo * 64 + 64] + alpha * s11;
}

// final: out[b,c,w,h] = res[b,h,w,c]
__global__ void k_final(const float* __restrict__ res, float* __restrict__ out) {
  long idx = (long)blockIdx.x * blockDim.x + threadIdx.x;
  long total = 8L * 64 * 128 * 128;
  if (idx >= total) return;
  int h = (int)(idx % 128);
  int w = (int)((idx / 128) % 128);
  int c = (int)((idx / (128 * 128)) % 64);
  int b = (int)(idx / (128L * 128 * 64));
  out[idx] = res[(((long)b * 16384) + h * 128 + w) * 64 + c];
}

// ================================ host side ==================================
static inline unsigned gdiv(long n, int b) { return (unsigned)((n + b - 1) / b); }

extern "C" void kernel_launch(void* const* d_in, const int* in_sizes, int n_in,
                              void* d_out, int out_size, void* d_ws, size_t ws_size,
                              hipStream_t stream) {
  (void)in_sizes; (void)n_in; (void)out_size; (void)ws_size;
  const float* x          = (const float*)d_in[0];
  const float* ln_w       = (const float*)d_in[1];
  const float* ln_b       = (const float*)d_in[2];
  const float* skip_scale = (const float*)d_in[3];
  const float* proj_w     = (const float*)d_in[4];
  const float* proj_b     = (const float*)d_in[5];
  const float* in_proj_w  = (const float*)d_in[6];
  const float* conv_w     = (const float*)d_in[7];
  const float* conv_b     = (const float*)d_in[8];
  const float* x_proj_w   = (const float*)d_in[9];
  const float* dt_proj_w  = (const float*)d_in[10];
  const float* dt_proj_b  = (const float*)d_in[11];
  const float* A_log      = (const float*)d_in[12];
  const float* Dv         = (const float*)d_in[13];
  const float* out_proj_w = (const float*)d_in[14];
  const float* down_w     = (const float*)d_in[15];
  const float* up_w       = (const float*)d_in[16];

  const int B = 8;
  int Hl[5], nl[5];
  long xh_off[5], out_off[5];
  long acc = 0;
  for (int l = 0; l < 5; ++l) {
    Hl[l] = 128 >> l;
    nl[l] = Hl[l] * Hl[l];
    xh_off[l] = acc;
    out_off[l] = acc;
    acc += (long)B * nl[l] * 64;
  }
  const long xh_total = acc;           // 11,173,888 floats
  const long T0 = 32L * 16384;

  float* ws = (float*)d_ws;
  long p = 0;
  float* xh    = ws + p; p += xh_total;
  float* outs  = ws + p; p += xh_total;
  float* xc    = ws + p; p += T0 * 16;
  float* xmraw = ws + p; p += T0 * 32;   // reused as dt after conv
  float* zbuf  = ws + p; p += T0 * 32;
  float* xm    = ws + p; p += T0 * 32;
  float* xdbl  = ws + p; p += T0 * 48;
  float* ys    = ws + p; p += T0 * 32;
  float* ym    = ws + p; p += T0 * 16;
  float* ymn   = ws + p; p += 8L * 16384 * 64;
  float* resa  = ws + p; p += 8L * 16384 * 64;
  float* resb  = ws + p; p += 8L * 16384 * 64;
  float* dtb_  = xmraw;                  // alias: xm_raw dead after conv

  // level-0 input layout change
  k_nchw_to_nhwc<<<gdiv(8L * 64 * 128 * 128, 256), 256, 0, stream>>>(
      x, xh + xh_off[0], B, 64, 128, 128);

  for (int l = 0; l < 5; ++l) {
    if (l > 0) {
      k_down<<<gdiv((long)B * nl[l] * 64, 256), 256, 0, stream>>>(
          xh + xh_off[l - 1], down_w, xh + xh_off[l], B, Hl[l - 1], Hl[l - 1]);
    }
    const int  n  = nl[l];
    const long T  = 32L * n;      // 4B sequences of length n
    const long BT = (long)B * n;
    float* xhl  = xh + xh_off[l];
    float* outl = outs + out_off[l];

    k_ln_split<<<gdiv(BT, 256), 256, 0, stream>>>(xhl, ln_w, ln_b, xc, B, n);

    const int tiles = (int)(T / 16);
    k_inproj<<<gdiv(tiles, 4), 128, 0, stream>>>(xc, in_proj_w, xmraw, zbuf, tiles);
    k_conv_silu<<<gdiv(T * 32, 256), 256, 0, stream>>>(xmraw, conv_w, conv_b, xm, T, n);
    k_xproj<<<gdiv(tiles, 4), 128, 0, stream>>>(xm, x_proj_w, xdbl, tiles);
    k_dt<<<gdiv(T * 32, 256), 256, 0, stream>>>(xdbl, dt_proj_w, dt_proj_b, dtb_, T);

    dim3 sgrid(32, 16);
    k_scan<<<sgrid, 32, 0, stream>>>(xm, dtb_, xdbl, A_log, ys, n);

    k_ycomb<<<gdiv(T * 32, 256), 256, 0, stream>>>(ys, xm, zbuf, Dv, T);
    k_outproj<<<gdiv(tiles, 4), 128, 0, stream>>>(ys, out_proj_w, xc, skip_scale, ym, tiles);
    k_merge_ln<<<gdiv(BT, 256), 256, 0, stream>>>(ym, ln_w, ln_b, ymn, B, n);

    const int ptiles = (int)(BT / 16);
    k_proj<<<gdiv(ptiles, 4), 128, 0, stream>>>(ymn, proj_w, proj_b, outl, ptiles);
  }

  // pyramid combine: res = out4; res = out_l + (1/(4-l)) * up(res), l = 3..0
  const float* cur = outs + out_off[4];
  float* pingpong[2] = {resa, resb};
  int pb = 0;
  for (int lev = 3; lev >= 0; --lev) {
    const float alpha = 1.0f / (float)(4 - lev);
    float* nxt = pingpong[pb];
    pb ^= 1;
    const int Hs = Hl[lev + 1];
    k_upadd<<<gdiv((long)B * Hs * Hs * 64, 256), 256, 0, stream>>>(
        cur, up_w, outs + out_off[lev], alpha, nxt, B, Hs, Hs);
    cur = nxt;
  }

  k_final<<<gdiv(8L * 64 * 128 * 128, 256), 256, 0, stream>>>(cur, (float*)d_out);
}
